// GINELayer_61950608277980
// MI455X (gfx1250) — compile-verified, matrix operand order checked by reading, not run on hardware
//
#include <hip/hip_runtime.h>

typedef __attribute__((ext_vector_type(16))) _Float16 v16h;
typedef __attribute__((ext_vector_type(8)))  _Float16 v8h;
typedef __attribute__((ext_vector_type(8)))  float    v8f;

#define BSZ 2
#define NN  512
#define HH  128

// ---------------- workspace layout ----------------
// [0,        512KB)  xm  : x*mask, f32   (B*N*H)
// [512KB,    768KB)  hn  : h_norm, f16   (B*N*H)
// [768KB,    832KB)  wsW : W1,W2 f16 pre-swizzled to WMMA A-operand layout
//                          (2 matrices * 8 ot * 4 kc * 32 lanes * 16 halves)

// ---------- kernel 1: mask + LayerNorm (f32 in, f32 xm + f16 h_norm out) ----
__global__ void gine_norm_kernel(const float* __restrict__ x,
                                 const float* __restrict__ mask,
                                 const float* __restrict__ gamma,
                                 const float* __restrict__ beta,
                                 float* __restrict__ xm,
                                 _Float16* __restrict__ hn)
{
    int row = blockIdx.x;            // b*N + n
    int t   = threadIdx.x;           // feature 0..127
    __shared__ float red[HH];
    float mk = mask[row];
    float xv = x[(size_t)row * HH + t] * mk;
    red[t] = xv; __syncthreads();
    for (int s = HH / 2; s > 0; s >>= 1) {
        if (t < s) red[t] += red[t + s];
        __syncthreads();
    }
    float mu = red[0] * (1.0f / HH);
    __syncthreads();
    float d = xv - mu;
    red[t] = d * d; __syncthreads();
    for (int s = HH / 2; s > 0; s >>= 1) {
        if (t < s) red[t] += red[t + s];
        __syncthreads();
    }
    float var = red[0] * (1.0f / HH);
    float hv  = d * rsqrtf(var + 1e-5f) * gamma[t] + beta[t];
    xm[(size_t)row * HH + t] = xv;
    hn[(size_t)row * HH + t] = (_Float16)hv;
}

// ---- kernel 2: swizzle W1/W2 (f32, row-major [o][h]) into f16 A-operand ----
// A-operand tile (ot,kc) covers rows o = ot*16+m (m=lane%16), k = kc*32 + kl
// lane layout: halves 0..7  -> k = (lane<16 ? 0 : 8)  + h
//              halves 8..15 -> k = (lane<16 ? 16: 24) + (h-8)
__global__ void gine_swizzleW_kernel(const float* __restrict__ W1,
                                     const float* __restrict__ W2,
                                     _Float16* __restrict__ wsW)
{
    int idx  = blockIdx.x * blockDim.x + threadIdx.x;   // 2048 total
    int lane = idx & 31;
    int kc   = (idx >> 5) & 3;
    int ot   = (idx >> 7) & 7;
    int mat  = (idx >> 10) & 1;
    const float* W = mat ? W2 : W1;
    _Float16* dst  = wsW + mat * 16384 + ((ot * 4 + kc) * 32 + lane) * 16;
    int m  = lane & 15;
    int lo = (lane & 16) ? 8 : 0;
#pragma unroll
    for (int h = 0; h < 16; ++h) {
        int kl = (h < 8) ? (lo + h) : (16 + lo + (h - 8));
        dst[h] = (_Float16)W[(ot * 16 + m) * HH + kc * 32 + kl];
    }
}

// ------------------------- kernel 3: main fused layer -----------------------
__global__ __launch_bounds__(256, 1)
void gine_main_kernel(const float*    __restrict__ edge,
                      const int*      __restrict__ adj,
                      const float*    __restrict__ mask,
                      const float*    __restrict__ b1,
                      const float*    __restrict__ b2,
                      const float*    __restrict__ xm,
                      const _Float16* __restrict__ hn,
                      const _Float16* __restrict__ wsW,
                      float*          __restrict__ out)
{
    __shared__ __align__(32) _Float16 sW[2 * 16384];     // 64KB swizzled weights
    __shared__ __align__(16) _Float16 sH1[8 * 8 * 32 * 8]; // 32KB h1 staging
    __shared__ float sB1[HH], sB2[HH];
    __shared__ float sPart[HH];

    int tid  = threadIdx.x;
    int lane = tid & 31;
    int w    = tid >> 5;
    int b    = blockIdx.x / NN;
    int i    = blockIdx.x % NN;

    {   // stage weights into LDS (64KB = 4096 uint4)
        const uint4* g = (const uint4*)wsW;
        uint4* s = (uint4*)sW;
        for (int idx = tid; idx < 4096; idx += 256) s[idx] = g[idx];
    }
    if (tid < HH) { sB1[tid] = b1[tid]; sB2[tid] = b2[tid]; sPart[tid] = 0.f; }
    __syncthreads();

    const int moff = (lane & 16) ? 8 : 0;   // feature sub-offset for lane half
    const int mrow = lane & 15;             // j-row owned by this lane
    const size_t rowbase = (size_t)(b * NN + i) * NN;

    for (int jt = w; jt < NN / 16; jt += 8) {
        int jrow = jt * 16 + mrow;
        float am = (float)adj[rowbase + jrow];
        const float*    erow = edge + (rowbase + jrow) * HH;
        const _Float16* hrow = hn + (size_t)(b * NN + jrow) * HH;

        // --- build B operand of GEMM1: m_in^T chunks (f32 edge + f16 hnorm) -
        v16h b1c[4];
#pragma unroll
        for (int kc = 0; kc < 4; ++kc) {
            int k0 = kc * 32 + moff;
            float ef[16];
            *(float4*)&ef[0]  = *(const float4*)(erow + k0);
            *(float4*)&ef[4]  = *(const float4*)(erow + k0 + 4);
            *(float4*)&ef[8]  = *(const float4*)(erow + k0 + 16);
            *(float4*)&ef[12] = *(const float4*)(erow + k0 + 20);
            v8h h0 = *(const v8h*)(hrow + k0);
            v8h h1 = *(const v8h*)(hrow + k0 + 16);
            v16h bb;
#pragma unroll
            for (int t2 = 0; t2 < 8; ++t2) {
                bb[t2]     = (_Float16)(ef[t2]     + (float)h0[t2]);
                bb[8 + t2] = (_Float16)(ef[8 + t2] + (float)h1[t2]);
            }
            b1c[kc] = bb;
        }

        // prefetch next j-tile's edge rows into cache (global_prefetch_b8)
        if (jt + 8 < NN / 16) {
            const float* enext = edge + (rowbase + (jt + 8) * 16 + mrow) * HH;
#pragma unroll
            for (int kc = 0; kc < 4; ++kc)
                __builtin_prefetch(enext + kc * 32, 0, 1);
        }

        // ---------------- GEMM1: D1^T = W1 x m_in^T ; stage h1 to LDS -------
#pragma unroll 1
        for (int ot = 0; ot < 8; ++ot) {
            v8f a1 = (v8f)0.f;
#pragma unroll
            for (int kc = 0; kc < 4; ++kc) {
                const v16h aw = *(const v16h*)&sW[((ot * 4 + kc) * 32 + lane) * 16];
                a1 = __builtin_amdgcn_wmma_f32_16x16x32_f16(
                    false, aw, false, b1c[kc], (short)0, a1, false, false);
            }
            float bvv[8];
            *(float4*)&bvv[0] = *(const float4*)&sB1[ot * 16 + moff];
            *(float4*)&bvv[4] = *(const float4*)&sB1[ot * 16 + moff + 4];
            v8h pk;
#pragma unroll
            for (int r = 0; r < 8; ++r)
                pk[r] = (_Float16)fmaxf(a1[r] + bvv[r], 0.f);
            *(v8h*)&sH1[((w * 8 + ot) * 32 + lane) * 8] = pk;
        }

        // --- rebuild B operand of GEMM2 from staged h1 (same-wave LDS RAW) --
        v16h b2c[4];
#pragma unroll
        for (int kc = 0; kc < 4; ++kc) {
            v8h lo = *(const v8h*)&sH1[((w * 8 + 2 * kc)     * 32 + lane) * 8];
            v8h hi = *(const v8h*)&sH1[((w * 8 + 2 * kc + 1) * 32 + lane) * 8];
            b2c[kc] = __builtin_shufflevector(lo, hi,
                0, 1, 2, 3, 4, 5, 6, 7, 8, 9, 10, 11, 12, 13, 14, 15);
        }

        // ---- GEMM2: D2^T = W2 x h1^T ; fused epilogue + LDS atomic reduce --
#pragma unroll 1
        for (int ot = 0; ot < 8; ++ot) {
            v8f a2 = (v8f)0.f;
#pragma unroll
            for (int kc = 0; kc < 4; ++kc) {
                const v16h aw =
                    *(const v16h*)&sW[16384 + ((ot * 4 + kc) * 32 + lane) * 16];
                a2 = __builtin_amdgcn_wmma_f32_16x16x32_f16(
                    false, aw, false, b2c[kc], (short)0, a2, false, false);
            }
            float bvv[8];
            *(float4*)&bvv[0] = *(const float4*)&sB2[ot * 16 + moff];
            *(float4*)&bvv[4] = *(const float4*)&sB2[ot * 16 + moff + 4];
#pragma unroll
            for (int r = 0; r < 8; ++r) {
                float v = fmaxf(a2[r] + bvv[r], 0.f) * am;
                // sum over the 16 j-rows held across each 16-lane group
                v += __shfl_xor(v, 1, 32);
                v += __shfl_xor(v, 2, 32);
                v += __shfl_xor(v, 4, 32);
                v += __shfl_xor(v, 8, 32);
                if ((lane & 15) == 0)
                    atomicAdd(&sPart[ot * 16 + moff + r], v);  // ds_add_f32
            }
        }
    }
    __syncthreads();

    if (tid < HH) {
        int row = b * NN + i;
        out[(size_t)row * HH + tid] =
            (xm[(size_t)row * HH + tid] + sPart[tid]) * mask[row];
    }
}

// ------------------------------- launcher -----------------------------------
extern "C" void kernel_launch(void* const* d_in, const int* in_sizes, int n_in,
                              void* d_out, int out_size, void* d_ws, size_t ws_size,
                              hipStream_t stream) {
    const float* x     = (const float*)d_in[0];
    const int*   adj   = (const int*)  d_in[1];
    const float* edge  = (const float*)d_in[2];
    const float* mask  = (const float*)d_in[3];
    const float* gamma = (const float*)d_in[4];
    const float* beta  = (const float*)d_in[5];
    const float* W1    = (const float*)d_in[6];
    const float* b1    = (const float*)d_in[7];
    const float* W2    = (const float*)d_in[8];
    const float* b2    = (const float*)d_in[9];
    float* out = (float*)d_out;

    char* ws = (char*)d_ws;
    float*     xm  = (float*)ws;
    _Float16*  hnp = (_Float16*)(ws + (size_t)BSZ * NN * HH * 4);
    _Float16*  wsW = (_Float16*)(ws + (size_t)BSZ * NN * HH * 4
                                    + (size_t)BSZ * NN * HH * 2);

    gine_norm_kernel<<<BSZ * NN, HH, 0, stream>>>(x, mask, gamma, beta, xm, hnp);
    gine_swizzleW_kernel<<<8, 256, 0, stream>>>(W1, W2, wsW);
    gine_main_kernel<<<BSZ * NN, 256, 0, stream>>>(edge, adj, mask, b1, b2,
                                                   xm, hnp, wsW, out);
}
